// HiqUnetKPCNMedium_50379966382790
// MI455X (gfx1250) — matrix operand order repaired and checked by reading, not compiled, not run on hardware
//
#include <hip/hip_runtime.h>
#include <hip/hip_bf16.h>
#include <math.h>

typedef __attribute__((ext_vector_type(16))) _Float16 v16h;
typedef __attribute__((ext_vector_type(8)))  _Float16 v8h;
typedef __attribute__((ext_vector_type(8)))  float    v8f;

#define LDS_RS 40   // padded LDS row stride in halfs (80 B, 16B-aligned, bank-friendly)

__device__ __forceinline__ v16h make_afrag(const _Float16* row, int ka)
{
    // A fragment (16x32 MxK, ISA 7.12.2): elem i -> k = (i&7) + ka + (i&8?16:0)
    v8h lo = *(const v8h*)(row + ka);        // 16B aligned (ka in {0,8})
    v8h hi = *(const v8h*)(row + ka + 16);
    v16h a;
    #pragma unroll
    for (int i = 0; i < 8; ++i) { a[i] = lo[i]; a[i + 8] = hi[i]; }
    return a;
}

__device__ __forceinline__ v16h make_bfrag(const _Float16* row, int kb)
{
    // B fragment (32x16 KxN): elem i -> k = (i&7) + kb/2 + (i&8?16:0) (kb = lane&16), col = row index
    v8h lo = *(const v8h*)(row + (kb >> 1));
    v8h hi = *(const v8h*)(row + (kb >> 1) + 16);
    v16h b;
    #pragma unroll
    for (int i = 0; i < 8; ++i) { b[i] = lo[i]; b[i + 8] = hi[i]; }
    return b;
}

// ---------------------------------------------------------------------------
// Implicit-GEMM conv (R=3 pad 1, or R=1) via v_wmma_f32_16x16x32_f16.
// Block = 256 thr = 8 waves. Tile: 64 couts x 64 pixels.
// Each wave: 2 WMMA M-subtiles x 1 N-subtile, double-buffered LDS.
// X: [Cin][H][W], Wt: [Cout][Cin][R][R], Y: [Cout][H][W]
// ---------------------------------------------------------------------------
template<int R, bool RELU>
__global__ __launch_bounds__(256) void conv_wmma(
    const float* __restrict__ X, const float* __restrict__ Wt,
    const float* __restrict__ Bias, float* __restrict__ Y,
    int Cin, int Cout, int H, int W)
{
    constexpr int RR  = R * R;
    constexpr int PAD = R / 2;
    const int HWn  = H * W;
    const int Ktot = Cin * RR;
    const int pbase = blockIdx.x * 64;
    const int mbase = blockIdx.y * 64;
    const int tid   = threadIdx.x;
    const int lane  = tid & 31;
    const int wave  = tid >> 5;
    const int waveN = wave & 3;   // 16-pixel subtile
    const int waveM = wave >> 2;  // 32-cout pair of subtiles

    __shared__ __align__(32) _Float16 lB[2][64][LDS_RS];  // [buf][pixel][k]
    __shared__ __align__(32) _Float16 lA[2][64][LDS_RS];  // [buf][cout][k]

    v8f acc0 = {}, acc1 = {};
    const int nsteps = (Ktot + 31) >> 5;

    // B staging: lanes walk pixels (n) -> coalesced global loads
    auto stageB = [&](int k0, int buf) {
        #pragma unroll
        for (int t = 0; t < 8; ++t) {
            int idx = tid + t * 256;
            int n = idx & 63, kk = idx >> 6;
            int k = k0 + kk;
            float v = 0.f;
            if (k < Ktot) {
                int c = k / RR, rs = k % RR;
                int r = rs / R, s = rs - r * R;
                int p = pbase + n;
                int h = p / W, w = p - h * W;
                int ih = h + r - PAD, iw = w + s - PAD;
                if (ih >= 0 && ih < H && iw >= 0 && iw < W)
                    v = X[(c * H + ih) * W + iw];
            }
            lB[buf][n][kk] = (_Float16)v;
        }
    };
    // A staging: weights are [Cout][K] with K contiguous -> lanes walk k
    auto stageA = [&](int k0, int buf) {
        #pragma unroll
        for (int t = 0; t < 8; ++t) {
            int idx = tid + t * 256;
            int m = idx >> 5, kk = idx & 31;
            int k = k0 + kk;
            int mg = mbase + m;
            float v = 0.f;
            if (k < Ktot && mg < Cout) {
                v = Wt[(size_t)mg * Ktot + k];
                if (k + 32 < Ktot)  // pull next K-step's weight line ahead
                    __builtin_prefetch(&Wt[(size_t)mg * Ktot + k + 32], 0, 1);
            }
            lA[buf][m][kk] = (_Float16)v;
        }
    };

    stageB(0, 0);
    stageA(0, 0);

    for (int s = 0; s < nsteps; ++s) {
        __syncthreads();                       // staging of buf s&1 visible
        if (s + 1 < nsteps) {                  // overlap: stage next buffer
            stageB((s + 1) * 32, (s + 1) & 1);
            stageA((s + 1) * 32, (s + 1) & 1);
        }
        const int buf = s & 1;
        const int ka  = (lane & 16) ? 8 : 0;
        const int mrow = waveM * 32 + (lane & 15);
        v16h a0 = make_afrag(&lA[buf][mrow][0], ka);
        v16h a1 = make_afrag(&lA[buf][mrow + 16][0], ka);

        const int ncol = waveN * 16 + (lane & 15);
        v16h bv = make_bfrag(&lB[buf][ncol][0], lane & 16);

        acc0 = __builtin_amdgcn_wmma_f32_16x16x32_f16(false, a0, false, bv,
                                                      (short)0, acc0, false, false);
        acc1 = __builtin_amdgcn_wmma_f32_16x16x32_f16(false, a1, false, bv,
                                                      (short)0, acc1, false, false);
    }

    // D layout: VGPR r -> M = r + (lane>=16 ? 8 : 0), N = lane&15
    const int p = pbase + waveN * 16 + (lane & 15);
    const int mHalf = (lane & 16) ? 8 : 0;
    #pragma unroll
    for (int r = 0; r < 8; ++r) {
        int m0 = mbase + waveM * 32 + mHalf + r;
        int m1 = m0 + 16;
        if (m0 < Cout) {
            float v = acc0[r] + Bias[m0];
            if (RELU) v = fmaxf(v, 0.f);
            Y[(size_t)m0 * HWn + p] = v;
        }
        if (m1 < Cout) {
            float v = acc1[r] + Bias[m1];
            if (RELU) v = fmaxf(v, 0.f);
            Y[(size_t)m1 * HWn + p] = v;
        }
    }
}

// ---------------------------------------------------------------------------
// ConvTranspose2d(k=2, s=2) as one GEMM: M = Cout*4 (weight [Cin][Cout][2][2]
// is K-major, M contiguous), K = Cin, N = H*W (input spatial).
// m = o*4 + p*2 + q  ->  Y[o][2i+p][2j+q].  M is a multiple of 64 here.
// ---------------------------------------------------------------------------
__global__ __launch_bounds__(256) void upconv_wmma(
    const float* __restrict__ X, const float* __restrict__ Wt,
    const float* __restrict__ Bias, float* __restrict__ Y,
    int Cin, int Cout, int H, int W)
{
    const int HWn  = H * W;
    const int M    = Cout * 4;
    const int Ktot = Cin;
    const int pbase = blockIdx.x * 64;
    const int mbase = blockIdx.y * 64;
    const int tid   = threadIdx.x;
    const int lane  = tid & 31;
    const int wave  = tid >> 5;
    const int waveN = wave & 3;
    const int waveM = wave >> 2;

    __shared__ __align__(32) _Float16 lB[2][64][LDS_RS];
    __shared__ __align__(32) _Float16 lA[2][64][LDS_RS];

    v8f acc0 = {}, acc1 = {};
    const int nsteps = (Ktot + 31) >> 5;

    // lanes walk pixels -> coalesced
    auto stageB = [&](int k0, int buf) {
        #pragma unroll
        for (int t = 0; t < 8; ++t) {
            int idx = tid + t * 256;
            int n = idx & 63, kk = idx >> 6;
            int k = k0 + kk;
            float v = (k < Ktot) ? X[(size_t)k * HWn + (pbase + n)] : 0.f;
            lB[buf][n][kk] = (_Float16)v;
        }
    };
    // weight is [K][M] with M contiguous -> lanes walk m (coalesced)
    auto stageA = [&](int k0, int buf) {
        #pragma unroll
        for (int t = 0; t < 8; ++t) {
            int idx = tid + t * 256;
            int m = idx & 63, kk = idx >> 6;
            int k = k0 + kk;
            int mg = mbase + m;
            float v = 0.f;
            if (k < Ktot && mg < M) {
                v = Wt[(size_t)k * M + mg];
                if (k + 32 < Ktot)
                    __builtin_prefetch(&Wt[(size_t)(k + 32) * M + mg], 0, 1);
            }
            lA[buf][m][kk] = (_Float16)v;
        }
    };

    stageB(0, 0);
    stageA(0, 0);

    for (int s = 0; s < nsteps; ++s) {
        __syncthreads();
        if (s + 1 < nsteps) {
            stageB((s + 1) * 32, (s + 1) & 1);
            stageA((s + 1) * 32, (s + 1) & 1);
        }
        const int buf = s & 1;
        const int ka  = (lane & 16) ? 8 : 0;
        const int mrow = waveM * 32 + (lane & 15);
        v16h a0 = make_afrag(&lA[buf][mrow][0], ka);
        v16h a1 = make_afrag(&lA[buf][mrow + 16][0], ka);

        const int ncol = waveN * 16 + (lane & 15);
        v16h bv = make_bfrag(&lB[buf][ncol][0], lane & 16);

        acc0 = __builtin_amdgcn_wmma_f32_16x16x32_f16(false, a0, false, bv,
                                                      (short)0, acc0, false, false);
        acc1 = __builtin_amdgcn_wmma_f32_16x16x32_f16(false, a1, false, bv,
                                                      (short)0, acc1, false, false);
    }

    const int p = pbase + waveN * 16 + (lane & 15);
    const int i = p / W, j = p - i * W;
    const int mHalf = (lane & 16) ? 8 : 0;
    #pragma unroll
    for (int r = 0; r < 8; ++r) {
        int mm[2];
        mm[0] = mbase + waveM * 32 + mHalf + r;
        mm[1] = mm[0] + 16;
        float av[2] = { acc0[r], acc1[r] };
        #pragma unroll
        for (int u = 0; u < 2; ++u) {
            int m = mm[u];
            if (m < M) {
                int o = m >> 2, pp = (m >> 1) & 1, q = m & 1;
                Y[(size_t)o * 4 * HWn + (size_t)(2 * i + pp) * (2 * W) + (2 * j + q)]
                    = av[u] + Bias[o];
            }
        }
    }
}

// ---------------------------------------------------------------------------
__global__ void maxpool2_kernel(const float* __restrict__ X, float* __restrict__ Y,
                                int C, int H, int W)
{
    int Ho = H >> 1, Wo = W >> 1;
    int idx = blockIdx.x * 256 + threadIdx.x;
    int total = C * Ho * Wo;
    if (idx >= total) return;
    int wo = idx % Wo; int t = idx / Wo; int ho = t % Ho; int c = t / Ho;
    const float* base = X + ((size_t)c * H + 2 * ho) * W + 2 * wo;
    Y[idx] = fmaxf(fmaxf(base[0], base[1]), fmaxf(base[W], base[W + 1]));
}

// softplus each of 441 channels, normalize by clipped sum (in place)
__global__ void softplus_norm_kernel(float* __restrict__ Rw, int HWn)
{
    int p = blockIdx.x * 256 + threadIdx.x;
    if (p >= HWn) return;
    float sum = 0.f;
    for (int k = 0; k < 441; ++k) {
        float x = Rw[(size_t)k * HWn + p];
        float sp = (x > 20.f) ? x : log1pf(__expf(x));
        Rw[(size_t)k * HWn + p] = sp;
        sum += sp;
    }
    float inv = 1.f / fmaxf(sum, 1e-6f);
    for (int k = 0; k < 441; ++k)
        Rw[(size_t)k * HWn + p] *= inv;
}

// out[c,h,w] = sum_k w[k,h,w] * noisy_padded[c, h+ky-10, w+kx-10]
__global__ void kpcn_apply_kernel(const float* __restrict__ Wn,
                                  const float* __restrict__ noisy,
                                  float* __restrict__ Out, int H, int W)
{
    int HWn = H * W;
    int idx = blockIdx.x * 256 + threadIdx.x;
    if (idx >= 3 * HWn) return;
    int c = idx / HWn; int p = idx - c * HWn;
    int h = p / W, w = p - h * W;
    float acc = 0.f;
    for (int ky = 0; ky < 21; ++ky) {
        int ih = h + ky - 10;
        if (ih < 0 || ih >= H) continue;
        for (int kx = 0; kx < 21; ++kx) {
            int iw = w + kx - 10;
            if (iw < 0 || iw >= W) continue;
            acc += Wn[(size_t)(ky * 21 + kx) * HWn + p] * noisy[(size_t)c * HWn + ih * W + iw];
        }
    }
    Out[idx] = acc;  // noisy + (out - noisy) == out
}

// ---------------------------------------------------------------------------
static void conv3(const float* x, const float* w, const float* b, float* y,
                  int Cin, int Cout, int H, int W, hipStream_t s)
{
    dim3 g((H * W) / 64, (Cout + 63) / 64);
    conv_wmma<3, true><<<g, 256, 0, s>>>(x, w, b, y, Cin, Cout, H, W);
}

static void conv1(const float* x, const float* w, const float* b, float* y,
                  int Cin, int Cout, int H, int W, hipStream_t s)
{
    dim3 g((H * W) / 64, (Cout + 63) / 64);
    conv_wmma<1, false><<<g, 256, 0, s>>>(x, w, b, y, Cin, Cout, H, W);
}

static void upconv(const float* x, const float* w, const float* b, float* y,
                   int Cin, int Cout, int H, int W, hipStream_t s)
{
    dim3 g((H * W) / 64, (Cout * 4) / 64);
    upconv_wmma<<<g, 256, 0, s>>>(x, w, b, y, Cin, Cout, H, W);
}

static void maxpool(const float* x, float* y, int C, int H, int W, hipStream_t s)
{
    int total = C * (H / 2) * (W / 2);
    maxpool2_kernel<<<(total + 255) / 256, 256, 0, s>>>(x, y, C, H, W);
}

extern "C" void kernel_launch(void* const* d_in, const int* in_sizes, int n_in,
                              void* d_out, int out_size, void* d_ws, size_t ws_size,
                              hipStream_t stream)
{
    (void)in_sizes; (void)n_in; (void)out_size; (void)ws_size;
    const float* noisy = (const float*)d_in[0];
    // flattened params (dict insertion order):
    // 1..4  e1{w1,b1,w2,b2}; 5..8 e2; 9..12 e3; 13..16 bn;
    // 17,18 up3_w,b; 19..22 dec3; 23,24 up2; 25..28 dec2;
    // 29,30 up1; 31..34 dec1; 35,36 out_w,out_b
    const float* P[37];
    for (int i = 0; i < 37; ++i) P[i] = (const float*)d_in[i];

    const int H1 = 256, W1 = 256, HW1 = H1 * W1;
    const int H2 = 128, W2 = 128, HW2 = H2 * W2;
    const int H3 = 64,  W3 = 64,  HW3 = H3 * W3;
    const int H4 = 32,  W4 = 32;
    const int b = 96;

    float* ws = (float*)d_ws;
    size_t off = 0;
    float* cat1 = ws + off; off += (size_t)2 * b * HW1;   // [u1 | x1]
    float* cat2 = ws + off; off += (size_t)4 * b * HW2;   // [u2 | x2]
    float* cat3 = ws + off; off += (size_t)8 * b * HW3;   // [u3 | x3]
    float* tmpA = ws + off; off += (size_t)b * HW1;       // ping
    float* tmpB = ws + off; off += (size_t)b * HW1;       // pong
    float* raw  = ws + off; off += (size_t)441 * HW1;     // kernel logits

    float* x1 = cat1 + (size_t)b * HW1;
    float* x2 = cat2 + (size_t)2 * b * HW2;
    float* x3 = cat3 + (size_t)4 * b * HW3;

    // ---------------- encoder ----------------
    conv3(noisy, P[1],  P[2],  tmpA, 3,     b,     H1, W1, stream);   // e1.c1
    conv3(tmpA,  P[3],  P[4],  x1,   b,     b,     H1, W1, stream);   // e1.c2 -> skip1
    maxpool(x1, tmpA, b, H1, W1, stream);
    conv3(tmpA,  P[5],  P[6],  tmpB, b,     2 * b, H2, W2, stream);   // e2.c1
    conv3(tmpB,  P[7],  P[8],  x2,   2 * b, 2 * b, H2, W2, stream);   // e2.c2 -> skip2
    maxpool(x2, tmpA, 2 * b, H2, W2, stream);
    conv3(tmpA,  P[9],  P[10], tmpB, 2 * b, 4 * b, H3, W3, stream);   // e3.c1
    conv3(tmpB,  P[11], P[12], x3,   4 * b, 4 * b, H3, W3, stream);   // e3.c2 -> skip3
    maxpool(x3, tmpA, 4 * b, H3, W3, stream);

    // ---------------- bottleneck ----------------
    conv3(tmpA, P[13], P[14], tmpB, 4 * b, 4 * b, H4, W4, stream);
    conv3(tmpB, P[15], P[16], tmpA, 4 * b, 4 * b, H4, W4, stream);    // bn -> tmpA

    // ---------------- decoder ----------------
    upconv(tmpA, P[17], P[18], cat3, 4 * b, 4 * b, H4, W4, stream);   // u3 -> cat3[:4b]
    conv3(cat3, P[19], P[20], tmpA, 8 * b, 4 * b, H3, W3, stream);    // dec3.c1
    conv3(tmpA, P[21], P[22], tmpB, 4 * b, 4 * b, H3, W3, stream);    // dec3.c2 -> d3

    upconv(tmpB, P[23], P[24], cat2, 4 * b, 2 * b, H3, W3, stream);   // u2 -> cat2[:2b]
    conv3(cat2, P[25], P[26], tmpA, 4 * b, 2 * b, H2, W2, stream);    // dec2.c1
    conv3(tmpA, P[27], P[28], tmpB, 2 * b, 2 * b, H2, W2, stream);    // dec2.c2 -> d2

    upconv(tmpB, P[29], P[30], cat1, 2 * b, b, H2, W2, stream);       // u1 -> cat1[:b]
    conv3(cat1, P[31], P[32], tmpA, 2 * b, b, H1, W1, stream);        // dec1.c1
    conv3(tmpA, P[33], P[34], tmpB, b, b, H1, W1, stream);            // dec1.c2 -> d1

    // ---------------- kernel prediction head ----------------
    conv1(tmpB, P[35], P[36], raw, b, 441, H1, W1, stream);           // 1x1 -> logits
    softplus_norm_kernel<<<(HW1 + 255) / 256, 256, 0, stream>>>(raw, HW1);
    kpcn_apply_kernel<<<(3 * HW1 + 255) / 256, 256, 0, stream>>>(raw, noisy,
                                                                 (float*)d_out, H1, W1);
}
